// CrossHyperedgeGen_25099788878239
// MI455X (gfx1250) — matrix operand order repaired and checked by reading, not compiled, not run on hardware
//
#include <hip/hip_runtime.h>
#include <hip/hip_bf16.h>
#include <math.h>

// ---------------------------------------------------------------------------
// Problem constants
// ---------------------------------------------------------------------------
#define D_DIM   256
#define E_DIM   512
#define H_HEADS 8
#define KE_TOP  128
#define B_BATCH 8
#define NA_ROWS 4096
#define NB_ROWS 2048
#define CTX_DIM 1024              // 4*D
#define ED      (E_DIM * D_DIM)   // 131072

typedef _Float16 v16h __attribute__((ext_vector_type(16)));
typedef _Float16 v8h  __attribute__((ext_vector_type(8)));
typedef float    v8f  __attribute__((ext_vector_type(8)));

union FragU { v16h v; v8h h[2]; };

// ---------------------------------------------------------------------------
// Kernel: f32 -> f16 conversion (grid-stride)
// ---------------------------------------------------------------------------
__global__ void cvt_f32_to_f16(const float* __restrict__ src,
                               _Float16* __restrict__ dst, int n) {
    for (int i = blockIdx.x * blockDim.x + threadIdx.x; i < n;
         i += gridDim.x * blockDim.x)
        dst[i] = (_Float16)src[i];
}

// ---------------------------------------------------------------------------
// Kernel: context reduction stage 1 (partial mean/max over row chunks)
// grid.x = B * nChunks, 256 threads (one per column d)
// ---------------------------------------------------------------------------
__global__ void ctx_partial(const float* __restrict__ X, int nChunks,
                            int rowsPerChunk,
                            float* __restrict__ psum, float* __restrict__ pmax) {
    const int b = blockIdx.x / nChunks;
    const int c = blockIdx.x % nChunks;
    const int d = threadIdx.x;
    const int N = nChunks * rowsPerChunk;
    const float* p = X + ((size_t)b * N + (size_t)c * rowsPerChunk) * D_DIM + d;
    float s = 0.0f, m = -3.402823e38f;
    for (int r = 0; r < rowsPerChunk; ++r) {
        float v = p[(size_t)r * D_DIM];
        s += v;
        m = fmaxf(m, v);
    }
    psum[(size_t)blockIdx.x * D_DIM + d] = s;
    pmax[(size_t)blockIdx.x * D_DIM + d] = m;
}

// ---------------------------------------------------------------------------
// Kernel: context reduction stage 2 -> ctx[b] = [meanA,maxA,meanB,maxB]
// grid = (2 srcs, 8 batches), 256 threads
// ---------------------------------------------------------------------------
__global__ void ctx_finish(const float* __restrict__ psumA, const float* __restrict__ pmaxA,
                           const float* __restrict__ psumB, const float* __restrict__ pmaxB,
                           float* __restrict__ ctx) {
    const int src = blockIdx.x;  // 0 = A, 1 = B
    const int b   = blockIdx.y;
    const int d   = threadIdx.x;
    const int  CH   = src ? 16 : 32;
    const float invN = src ? (1.0f / NB_ROWS) : (1.0f / NA_ROWS);
    const float* ps = (src ? psumB : psumA) + (size_t)b * CH * D_DIM;
    const float* pm = (src ? pmaxB : pmaxA) + (size_t)b * CH * D_DIM;
    float s = 0.0f, m = -3.402823e38f;
    for (int c = 0; c < CH; ++c) {
        s += ps[(size_t)c * D_DIM + d];
        m = fmaxf(m, pm[(size_t)c * D_DIM + d]);
    }
    ctx[(size_t)b * CTX_DIM + src * 512 + d]       = s * invN;
    ctx[(size_t)b * CTX_DIM + src * 512 + 256 + d] = m;
}

// ---------------------------------------------------------------------------
// Kernel: prototypes P_f16[b][o] = f16(base[o] + bc[o] + ctx[b] . Wc[o,:])
// One wave per Wc row (coalesced float4 streaming of the 512 MB weight,
// read exactly once; all 8 batches computed from LDS-resident ctx).
// grid.x = ED/8 blocks, 256 threads (8 waves)
// ---------------------------------------------------------------------------
__global__ void proto_kernel(const float* __restrict__ Wc,
                             const float* __restrict__ ctx,
                             const float* __restrict__ base,
                             const float* __restrict__ bc,
                             _Float16* __restrict__ P) {
    __shared__ float sctx[B_BATCH][CTX_DIM];
    const int t = threadIdx.x;
    for (int i = t; i < B_BATCH * CTX_DIM; i += 256)
        (&sctx[0][0])[i] = ctx[i];
    __syncthreads();

    const int wave = t >> 5, lane = t & 31;
    const long o = (long)blockIdx.x * 8 + wave;  // Wc row == flat (e*D+d)
    const float4* wrow = (const float4*)(Wc + (size_t)o * CTX_DIM);

    float acc[B_BATCH];
#pragma unroll
    for (int b = 0; b < B_BATCH; ++b) acc[b] = 0.0f;

#pragma unroll
    for (int it = 0; it < 8; ++it) {
        float4 w = wrow[it * 32 + lane];
        const int k0 = it * 128 + lane * 4;
#pragma unroll
        for (int b = 0; b < B_BATCH; ++b) {
            const float* cb = &sctx[b][k0];
            acc[b] += w.x * cb[0] + w.y * cb[1] + w.z * cb[2] + w.w * cb[3];
        }
    }
#pragma unroll
    for (int b = 0; b < B_BATCH; ++b)
#pragma unroll
        for (int m = 16; m >= 1; m >>= 1)
            acc[b] += __shfl_xor(acc[b], m, 32);

    if (lane == 0) {
        const float add = base[o] + bc[o];
#pragma unroll
        for (int b = 0; b < B_BATCH; ++b)
            P[(size_t)b * ED + o] = (_Float16)(add + acc[b]);
    }
}

// ---------------------------------------------------------------------------
// Kernel: NT GEMM via v_wmma_f32_16x16x32_f16.
//   C[m][n] = (sum_k A16[m,k] * B16[n,k]) * scale (+ bias[n])
// A, B are f16 row-major with leading dim K=256. Per the CDNA5 fragment
// tables, NT fragments are contiguous 16B per-lane loads (no LDS needed).
// Block = 256 threads (8 waves); block tile 128x128; wave tile 32x64.
// ---------------------------------------------------------------------------
template <typename TOut>
__global__ void gemm_nt_f16(const _Float16* __restrict__ A,
                            const _Float16* __restrict__ B,
                            TOut* __restrict__ C,
                            const float* __restrict__ bias,
                            float scale, int ldc,
                            long aBatch, long bBatch, long cBatch) {
    const int t    = threadIdx.x;
    const int wave = t >> 5, lane = t & 31;
    const int m16  = lane & 15;   // row (A) / col (B) / col (C) within 16-tile
    const int h    = lane >> 4;   // lane half -> K-half (A,B) / M-half (C)

    A += (size_t)blockIdx.z * aBatch;
    B += (size_t)blockIdx.z * bBatch;
    C += (size_t)blockIdx.z * cBatch;

    const int rowBase = blockIdx.x * 128 + (wave & 3) * 32;
    const int colBase = blockIdx.y * 128 + (wave >> 2) * 64;

    // A frag lane base: lane m+16h reads row m, f16 elems [8h..8h+7] and [16+8h..]
    const _Float16* a0 = A + (size_t)(rowBase + m16) * D_DIM + h * 8;
    const _Float16* a1 = a0 + 16 * D_DIM;
    // B frag lane base: lane n+16h reads row n, f16 elems [8h..8h+7] and [16+8h..]
    const _Float16* bp0 = B + (size_t)(colBase + m16) * D_DIM + h * 8;

    v8f acc[2][4];
#pragma unroll
    for (int i = 0; i < 2; ++i)
#pragma unroll
        for (int j = 0; j < 4; ++j) acc[i][j] = (v8f)(0.0f);

#pragma unroll
    for (int ks = 0; ks < 8; ++ks) {
        const int K0 = ks * 32;
        FragU a[2], bb[4];
        a[0].h[0] = *(const v8h*)(a0 + K0);
        a[0].h[1] = *(const v8h*)(a0 + K0 + 16);
        a[1].h[0] = *(const v8h*)(a1 + K0);
        a[1].h[1] = *(const v8h*)(a1 + K0 + 16);
#pragma unroll
        for (int j = 0; j < 4; ++j) {
            const _Float16* bpj = bp0 + (size_t)(16 * j) * D_DIM + K0;
            bb[j].h[0] = *(const v8h*)(bpj);
            bb[j].h[1] = *(const v8h*)(bpj + 16);
        }
#pragma unroll
        for (int i = 0; i < 2; ++i)
#pragma unroll
            for (int j = 0; j < 4; ++j)
                acc[i][j] = __builtin_amdgcn_wmma_f32_16x16x32_f16(
                    false, a[i].v, false, bb[j].v, (short)0, acc[i][j],
                    false, false);
    }

    // C/D layout: lane n+16*mh, VGPR v -> row (v + 8*mh), col n
#pragma unroll
    for (int j = 0; j < 4; ++j) {
        const int col = colBase + 16 * j + m16;
        const float badd = bias ? bias[col] : 0.0f;
#pragma unroll
        for (int i = 0; i < 2; ++i) {
            const int row0 = rowBase + 16 * i + 8 * h;
            TOut* cp = C + (size_t)row0 * ldc + col;
#pragma unroll
            for (int v = 0; v < 8; ++v)
                cp[(size_t)v * ldc] = (TOut)(acc[i][j][v] * scale + badd);
        }
    }
}

// ---------------------------------------------------------------------------
// Kernel: per-row top-128-of-512 + softmax.
// One block (256 threads) per row; bitonic sort of packed u64 keys in LDS.
// Key = (~sortable(value) << 32) | index  => ascending sort gives values
// descending with ties broken by lower index (matches jax.lax.top_k).
// ---------------------------------------------------------------------------
__global__ void topk_softmax(const float* __restrict__ logits,
                             int* __restrict__ outIdx,
                             float* __restrict__ outW) {
    __shared__ unsigned long long skey[E_DIM];
    __shared__ float sred[256];
    const int t = threadIdx.x;
    const size_t row = blockIdx.x;
    const float* lr = logits + row * E_DIM;

    for (int e = t; e < E_DIM; e += 256) {
        float v = lr[e];
        unsigned bits = __float_as_uint(v);
        unsigned s = bits ^ ((bits & 0x80000000u) ? 0xFFFFFFFFu : 0x80000000u);
        skey[e] = ((unsigned long long)(~s) << 32) | (unsigned)e;
    }
    __syncthreads();

    for (unsigned k = 2; k <= E_DIM; k <<= 1) {
        for (unsigned j = k >> 1; j > 0; j >>= 1) {
            for (unsigned i = t; i < E_DIM; i += 256) {
                unsigned ixj = i ^ j;
                if (ixj > i) {
                    bool up = ((i & k) == 0);
                    unsigned long long x = skey[i], y = skey[ixj];
                    if ((x > y) == up) { skey[i] = y; skey[ixj] = x; }
                }
            }
            __syncthreads();
        }
    }

    float v = 0.0f;
    unsigned idx = 0;
    if (t < KE_TOP) {
        unsigned long long kk = skey[t];
        idx = (unsigned)(kk & 0xFFFFFFFFu);
        unsigned s = ~(unsigned)(kk >> 32);
        unsigned bits = (s & 0x80000000u) ? (s ^ 0x80000000u) : ~s;
        v = __uint_as_float(bits);
        sred[t] = v;
    }
    __syncthreads();
    const float vmax = sred[0];  // sorted descending -> element 0 is max
    float ex = 0.0f;
    __syncthreads();
    if (t < KE_TOP) {
        ex = expf(v - vmax);
        sred[t] = ex;
    }
    __syncthreads();
    for (int st = 64; st > 0; st >>= 1) {
        if (t < st) sred[t] += sred[t + st];
        __syncthreads();
    }
    const float inv = 1.0f / sred[0];
    if (t < KE_TOP) {
        outIdx[row * KE_TOP + t] = (int)idx;
        outW[row * KE_TOP + t]   = ex * inv;
    }
}

// ---------------------------------------------------------------------------
// Launcher
// ---------------------------------------------------------------------------
extern "C" void kernel_launch(void* const* d_in, const int* in_sizes, int n_in,
                              void* d_out, int out_size, void* d_ws, size_t ws_size,
                              hipStream_t stream) {
    (void)in_sizes; (void)n_in; (void)out_size; (void)ws_size;
    const float* X_A  = (const float*)d_in[0];
    const float* X_B  = (const float*)d_in[1];
    const float* base = (const float*)d_in[2];
    const float* Wc   = (const float*)d_in[3];
    const float* bc   = (const float*)d_in[4];
    const float* WA   = (const float*)d_in[5];
    const float* bA   = (const float*)d_in[6];
    const float* WB   = (const float*)d_in[7];
    const float* bB   = (const float*)d_in[8];

    // ---- workspace carve-out (256B aligned) ----
    char* w = (char*)d_ws;
    size_t off = 0;
    auto alloc = [&](size_t bytes) {
        void* p = w + off;
        off = (off + bytes + 255) & ~(size_t)255;
        return p;
    };
    float*    psumA  = (float*)   alloc((size_t)B_BATCH * 32 * D_DIM * 4);
    float*    pmaxA  = (float*)   alloc((size_t)B_BATCH * 32 * D_DIM * 4);
    float*    psumB  = (float*)   alloc((size_t)B_BATCH * 16 * D_DIM * 4);
    float*    pmaxB  = (float*)   alloc((size_t)B_BATCH * 16 * D_DIM * 4);
    float*    ctx    = (float*)   alloc((size_t)B_BATCH * CTX_DIM * 4);
    _Float16* XhA    = (_Float16*)alloc((size_t)B_BATCH * NA_ROWS * D_DIM * 2);
    _Float16* XhB    = (_Float16*)alloc((size_t)B_BATCH * NB_ROWS * D_DIM * 2);
    _Float16* WhA    = (_Float16*)alloc((size_t)D_DIM * D_DIM * 2);
    _Float16* WhB    = (_Float16*)alloc((size_t)D_DIM * D_DIM * 2);
    _Float16* XpA    = (_Float16*)alloc((size_t)B_BATCH * NA_ROWS * D_DIM * 2);
    _Float16* XpB    = (_Float16*)alloc((size_t)B_BATCH * NB_ROWS * D_DIM * 2);
    _Float16* Ph     = (_Float16*)alloc((size_t)B_BATCH * ED * 2);
    float*    logA   = (float*)   alloc((size_t)B_BATCH * NA_ROWS * E_DIM * 4);
    float*    logB   = (float*)   alloc((size_t)B_BATCH * NB_ROWS * E_DIM * 4);

    // ---- outputs: topiA | wA | topiB | wB (4-byte slots, ints as raw bits) ----
    int*   topiA = (int*)d_out;
    float* wA    = (float*)d_out + (size_t)B_BATCH * NA_ROWS * KE_TOP;
    int*   topiB = (int*)((float*)d_out + 2ull * B_BATCH * NA_ROWS * KE_TOP);
    float* wB    = (float*)d_out + 2ull * B_BATCH * NA_ROWS * KE_TOP
                                 + (size_t)B_BATCH * NB_ROWS * KE_TOP;

    // 1) f32 -> f16 conversions
    cvt_f32_to_f16<<<4096, 256, 0, stream>>>(X_A, XhA, B_BATCH * NA_ROWS * D_DIM);
    cvt_f32_to_f16<<<4096, 256, 0, stream>>>(X_B, XhB, B_BATCH * NB_ROWS * D_DIM);
    cvt_f32_to_f16<<<256,  256, 0, stream>>>(WA, WhA, D_DIM * D_DIM);
    cvt_f32_to_f16<<<256,  256, 0, stream>>>(WB, WhB, D_DIM * D_DIM);

    // 2) context mean/max (two-stage)
    ctx_partial<<<B_BATCH * 32, 256, 0, stream>>>(X_A, 32, 128, psumA, pmaxA);
    ctx_partial<<<B_BATCH * 16, 256, 0, stream>>>(X_B, 16, 128, psumB, pmaxB);
    ctx_finish<<<dim3(2, B_BATCH), 256, 0, stream>>>(psumA, pmaxA, psumB, pmaxB, ctx);

    // 3) prototypes: base + bc + ctx @ Wc.T   (streams the 512 MB Wc once)
    proto_kernel<<<ED / 8, 256, 0, stream>>>(Wc, ctx, base, bc, Ph);

    // 4) projections: Xp = f16(X @ W.T + b)   (WMMA, f32 accumulate)
    gemm_nt_f16<_Float16><<<dim3((B_BATCH * NA_ROWS) / 128, D_DIM / 128, 1), 256, 0, stream>>>(
        XhA, WhA, XpA, bA, 1.0f, D_DIM, 0, 0, 0);
    gemm_nt_f16<_Float16><<<dim3((B_BATCH * NB_ROWS) / 128, D_DIM / 128, 1), 256, 0, stream>>>(
        XhB, WhB, XpB, bB, 1.0f, D_DIM, 0, 0, 0);

    // 5) logits = (Xp @ P.T) / (sqrt(DH) * H)   (WMMA, per-batch B matrix)
    const float LSCALE = 0.02209708691207961f;  // 1 / (8 * sqrt(32))
    gemm_nt_f16<float><<<dim3(NA_ROWS / 128, E_DIM / 128, B_BATCH), 256, 0, stream>>>(
        XpA, Ph, logA, nullptr, LSCALE, E_DIM,
        (long)NA_ROWS * D_DIM, (long)ED, (long)NA_ROWS * E_DIM);
    gemm_nt_f16<float><<<dim3(NB_ROWS / 128, E_DIM / 128, B_BATCH), 256, 0, stream>>>(
        XpB, Ph, logB, nullptr, LSCALE, E_DIM,
        (long)NB_ROWS * D_DIM, (long)ED, (long)NB_ROWS * E_DIM);

    // 6) top-k + softmax per row
    topk_softmax<<<B_BATCH * NA_ROWS, 256, 0, stream>>>(logA, topiA, wA);
    topk_softmax<<<B_BATCH * NB_ROWS, 256, 0, stream>>>(logB, topiB, wB);
}
